// MultiHeadAttention_86775519248667
// MI455X (gfx1250) — compile-verified, matrix operand order checked by reading, not run on hardware
//
#include <hip/hip_runtime.h>

typedef _Float16 half_t;
typedef __attribute__((ext_vector_type(16))) _Float16 v16h;
typedef __attribute__((ext_vector_type(2)))  _Float16 h2;
typedef __attribute__((ext_vector_type(8)))  float    v8f;

constexpr int Bc = 2, Tc = 2048, Cc = 2048, Hc = 16, Dc = 128;
constexpr int Mc = Bc * Tc;                       // 4096 rows of x
constexpr float SM_SCALE = 0.08838834764831845f;  // 1/sqrt(128)

// Element e (0..15) of a 16-lane-group fragment maps to K index:
// VGPR v=e/2 holds K pairs; v0..3 -> K=0..7, v4..7 -> K=16..23, +8 for upper half-wave.
__device__ __forceinline__ int kmap(int e, int hg) {
  return ((e & 8) << 1) + (e & 7) + (hg << 3);
}

// ---------------------------------------------------------------------------
// GEMM: Out = A(MxK, K-contig) * W(NxK, K-contig)^T + bias   (f32 accumulate)
// Block tile 128x128, 8 waves of 64x32, K staged 64 at a time through LDS.
// A_HALF: A is f16 (workspace) else f32 (input x).
// HEAD_OUT: write f16 to [B,H,T,D] layout, else f32 row-major [M,N].
// ---------------------------------------------------------------------------
template <bool A_HALF, bool HEAD_OUT>
__global__ __launch_bounds__(256)
void gemm_nt_bias(const void* __restrict__ Aptr,
                  const float* __restrict__ W,
                  const float* __restrict__ bias,
                  void* __restrict__ Outptr)
{
  __shared__ half_t As[128][72];   // 128 rows x 64 K (+8 pad)
  __shared__ half_t Bs[128][72];

  const int lane = threadIdx.x & 31;
  const int wave = threadIdx.x >> 5;
  const int mw   = wave >> 2;          // 0..1  (64-row slab)
  const int nw   = wave & 3;           // 0..3  (32-col slab)
  const int hg   = lane >> 4;          // half-wave group
  const int lm   = lane & 15;

  const int mBlock = blockIdx.y * 128;
  const int nBlock = blockIdx.x * 128;

  v8f acc[4][2] = {};

  const int ldr = threadIdx.x >> 1;         // 0..127
  const int ldc = (threadIdx.x & 1) * 32;   // 0 or 32

  for (int k0 = 0; k0 < Cc; k0 += 64) {
    __syncthreads();
    // ---- stage A tile (128 x 64) into LDS as f16 ----
    if constexpr (A_HALF) {
      const half_t* src = (const half_t*)Aptr + (size_t)(mBlock + ldr) * Cc + k0 + ldc;
      #pragma unroll
      for (int j = 0; j < 32; j += 8)
        *(uint4*)&As[ldr][ldc + j] = ((const uint4*)src)[j >> 3];
      __builtin_prefetch(src + 64, 0, 0);
    } else {
      const float* src = (const float*)Aptr + (size_t)(mBlock + ldr) * Cc + k0 + ldc;
      #pragma unroll
      for (int j = 0; j < 32; j += 4) {
        float4 f = *(const float4*)(src + j);
        As[ldr][ldc + j + 0] = (half_t)f.x;
        As[ldr][ldc + j + 1] = (half_t)f.y;
        As[ldr][ldc + j + 2] = (half_t)f.z;
        As[ldr][ldc + j + 3] = (half_t)f.w;
      }
      __builtin_prefetch(src + 64, 0, 0);
    }
    // ---- stage B tile (128 x 64) ----
    {
      const float* src = W + (size_t)(nBlock + ldr) * Cc + k0 + ldc;
      #pragma unroll
      for (int j = 0; j < 32; j += 4) {
        float4 f = *(const float4*)(src + j);
        Bs[ldr][ldc + j + 0] = (half_t)f.x;
        Bs[ldr][ldc + j + 1] = (half_t)f.y;
        Bs[ldr][ldc + j + 2] = (half_t)f.z;
        Bs[ldr][ldc + j + 3] = (half_t)f.w;
      }
      __builtin_prefetch(src + 64, 0, 0);
    }
    __syncthreads();

    #pragma unroll
    for (int c = 0; c < 2; ++c) {
      // ---- gather fragments for this 32-wide K chunk ----
      v16h af[4];
      #pragma unroll
      for (int mi = 0; mi < 4; ++mi) {
        const half_t* arow = &As[mw * 64 + mi * 16 + lm][c * 32];
        #pragma unroll
        for (int e = 0; e < 16; e += 2) {
          h2 p = *(const h2*)(arow + kmap(e, hg));
          af[mi][e] = p.x; af[mi][e + 1] = p.y;
        }
      }
      v16h bf[2];
      #pragma unroll
      for (int ni = 0; ni < 2; ++ni) {
        const half_t* brow = &Bs[nw * 32 + ni * 16 + lm][c * 32];
        #pragma unroll
        for (int e = 0; e < 16; e += 2) {
          h2 p = *(const h2*)(brow + kmap(e, hg));
          bf[ni][e] = p.x; bf[ni][e + 1] = p.y;
        }
      }
      #pragma unroll
      for (int mi = 0; mi < 4; ++mi)
        #pragma unroll
        for (int ni = 0; ni < 2; ++ni)
          acc[mi][ni] = __builtin_amdgcn_wmma_f32_16x16x32_f16(
              false, af[mi], false, bf[ni], (short)0, acc[mi][ni], false, false);
    }
  }

  // ---- epilogue: bias + store ----
  #pragma unroll
  for (int ni = 0; ni < 2; ++ni) {
    const int nAbs = nBlock + nw * 32 + ni * 16 + lm;
    const float bv = bias[nAbs];
    #pragma unroll
    for (int mi = 0; mi < 4; ++mi) {
      #pragma unroll
      for (int r = 0; r < 8; ++r) {
        const int mAbs = mBlock + mw * 64 + mi * 16 + r + 8 * hg;
        const float v = acc[mi][ni][r] + bv;
        if constexpr (HEAD_OUT) {
          const int bb = mAbs >> 11, tt = mAbs & (Tc - 1);
          const int hh = nAbs >> 7,  dd = nAbs & (Dc - 1);
          ((half_t*)Outptr)[((size_t)(bb * Hc + hh) * Tc + tt) * Dc + dd] = (half_t)v;
        } else {
          ((float*)Outptr)[(size_t)mAbs * Cc + nAbs] = v;
        }
      }
    }
  }
}

// ---------------------------------------------------------------------------
// RoPE on Q and K in place ([B,H,T,D] f16), f32 math.
// ---------------------------------------------------------------------------
__global__ __launch_bounds__(256)
void rope_qk(half_t* __restrict__ Q, half_t* __restrict__ Kv)
{
  const int idx = blockIdx.x * 256 + threadIdx.x;   // one (bh,t,d<64) pair
  const int d = idx & 63;
  const int t = (idx >> 6) & (Tc - 1);
  const size_t base = (size_t)(idx >> 6) * Dc;

  // inv_freq = theta^(-2d/D) = exp(-d * ln(10000)/64)
  const float ang = (float)t * expf(-(float)d * 0.14391157f);
  const float s = sinf(ang), c = cosf(ang);

  const float q0 = (float)Q[base + d],  q1 = (float)Q[base + d + 64];
  Q[base + d]      = (half_t)(q0 * c - q1 * s);
  Q[base + d + 64] = (half_t)(q1 * c + q0 * s);
  const float k0 = (float)Kv[base + d], k1 = (float)Kv[base + d + 64];
  Kv[base + d]      = (half_t)(k0 * c - k1 * s);
  Kv[base + d + 64] = (half_t)(k1 * c + k0 * s);
}

// ---------------------------------------------------------------------------
// Causal flash attention. Grid: (T/128, B*H). 8 waves x 16 queries per block.
// Q,K,V f16 [B,H,T,D]; out f16 [B,T,C] (ready for O-projection GEMM).
// V is staged TRANSPOSED in LDS so PV fragment gathers are contiguous h2 loads.
// ---------------------------------------------------------------------------
__global__ __launch_bounds__(256)
void flash_attn(const half_t* __restrict__ Q,
                const half_t* __restrict__ Kt,
                const half_t* __restrict__ V,
                half_t* __restrict__ AO)
{
  __shared__ half_t Ks[32][136];     // [key][d]
  __shared__ half_t Vt[128][40];     // [d][key]  (transposed)
  __shared__ half_t Ps[8][16][40];   // per-wave P staging

  const int lane = threadIdx.x & 31;
  const int wave = threadIdx.x >> 5;
  const int hg   = lane >> 4;
  const int lm   = lane & 15;

  const int bh = blockIdx.y;
  const int b  = bh >> 4;    // H = 16
  const int h  = bh & 15;
  const size_t headBase = (size_t)bh * Tc * Dc;

  const int qbase = blockIdx.x * 128 + wave * 16;

  // Q fragments, pre-scaled by 1/sqrt(D)
  v16h qf[4];
  {
    const half_t* qrow = Q + headBase + (size_t)(qbase + lm) * Dc;
    #pragma unroll
    for (int c = 0; c < 4; ++c)
      #pragma unroll
      for (int e = 0; e < 16; ++e)
        qf[c][e] = (half_t)((float)qrow[c * 32 + kmap(e, hg)] * SM_SCALE);
  }

  v8f o[8] = {};
  float mrun[8], lrun[8];
  #pragma unroll
  for (int r = 0; r < 8; ++r) { mrun[r] = -__builtin_inff(); lrun[r] = 0.f; }

  const int kbEnd = blockIdx.x * 4 + 4;       // key blocks of 32
  const int ldr = threadIdx.x >> 3;           // 0..31 key row
  const int ldc = (threadIdx.x & 7) * 16;     // 0..112 d base

  for (int kb = 0; kb < kbEnd; ++kb) {
    __syncthreads();
    {
      const half_t* ksrc = Kt + headBase + (size_t)(kb * 32 + ldr) * Dc + ldc;
      const half_t* vsrc = V  + headBase + (size_t)(kb * 32 + ldr) * Dc + ldc;
      *(uint4*)&Ks[ldr][ldc]     = ((const uint4*)ksrc)[0];
      *(uint4*)&Ks[ldr][ldc + 8] = ((const uint4*)ksrc)[1];
      half_t vtmp[16];
      *(uint4*)&vtmp[0] = ((const uint4*)vsrc)[0];
      *(uint4*)&vtmp[8] = ((const uint4*)vsrc)[1];
      #pragma unroll
      for (int j = 0; j < 16; ++j)
        Vt[ldc + j][ldr] = vtmp[j];           // transpose into LDS
      __builtin_prefetch(ksrc + 32 * Dc, 0, 0);
      __builtin_prefetch(vsrc + 32 * Dc, 0, 0);
    }
    __syncthreads();

    if (kb * 32 <= qbase) {   // causal skip; every kept row has >=1 live key
      // ---- scores: S = Qs * K^T over D = 4 chunks of 32 ----
      v8f s0 = {}, s1 = {};
      #pragma unroll
      for (int c = 0; c < 4; ++c) {
        v16h kf0, kf1;
        const half_t* kr0 = &Ks[lm][c * 32];
        const half_t* kr1 = &Ks[lm + 16][c * 32];
        #pragma unroll
        for (int e = 0; e < 16; e += 2) {
          const int kk = kmap(e, hg);
          h2 p0 = *(const h2*)(kr0 + kk);
          h2 p1 = *(const h2*)(kr1 + kk);
          kf0[e] = p0.x; kf0[e + 1] = p0.y;
          kf1[e] = p1.x; kf1[e + 1] = p1.y;
        }
        s0 = __builtin_amdgcn_wmma_f32_16x16x32_f16(false, qf[c], false, kf0, (short)0, s0, false, false);
        s1 = __builtin_amdgcn_wmma_f32_16x16x32_f16(false, qf[c], false, kf1, (short)0, s1, false, false);
      }

      // ---- online softmax (rows m = r + 8*hg, cols across 16 lanes) ----
      float alpha[8];
      #pragma unroll
      for (int r = 0; r < 8; ++r) {
        const int q = qbase + r + 8 * hg;
        const int key0 = kb * 32 + lm;
        const float a   = (key0      <= q) ? s0[r] : -__builtin_inff();
        const float bb2 = (key0 + 16 <= q) ? s1[r] : -__builtin_inff();
        float mx = fmaxf(a, bb2);
        mx = fmaxf(mx, __shfl_xor(mx, 1, 32));
        mx = fmaxf(mx, __shfl_xor(mx, 2, 32));
        mx = fmaxf(mx, __shfl_xor(mx, 4, 32));
        mx = fmaxf(mx, __shfl_xor(mx, 8, 32));
        const float mnew = fmaxf(mrun[r], mx);
        alpha[r] = __expf(mrun[r] - mnew);
        const float p0 = __expf(a - mnew);
        const float p1 = __expf(bb2 - mnew);
        float rs = p0 + p1;
        rs += __shfl_xor(rs, 1, 32);
        rs += __shfl_xor(rs, 2, 32);
        rs += __shfl_xor(rs, 4, 32);
        rs += __shfl_xor(rs, 8, 32);
        lrun[r] = lrun[r] * alpha[r] + rs;
        mrun[r] = mnew;
        Ps[wave][r + 8 * hg][lm]      = (half_t)p0;
        Ps[wave][r + 8 * hg][lm + 16] = (half_t)p1;
      }
      #pragma unroll
      for (int ni = 0; ni < 8; ++ni)
        #pragma unroll
        for (int r = 0; r < 8; ++r)
          o[ni][r] *= alpha[r];

      __builtin_amdgcn_wave_barrier();   // order Ps writes vs cross-lane reads

      // ---- P fragment (16x32) ----
      v16h pf;
      {
        const half_t* prow = &Ps[wave][lm][0];
        #pragma unroll
        for (int e = 0; e < 16; e += 2) {
          h2 p = *(const h2*)(prow + kmap(e, hg));
          pf[e] = p.x; pf[e + 1] = p.y;
        }
      }
      // ---- O += P * V (8 tiles across D); V^T rows are key-contiguous ----
      #pragma unroll
      for (int ni = 0; ni < 8; ++ni) {
        v16h vf;
        const half_t* vrow = &Vt[ni * 16 + lm][0];
        #pragma unroll
        for (int e = 0; e < 16; e += 2) {
          h2 p = *(const h2*)(vrow + kmap(e, hg));
          vf[e] = p.x; vf[e + 1] = p.y;
        }
        o[ni] = __builtin_amdgcn_wmma_f32_16x16x32_f16(false, pf, false, vf, (short)0, o[ni], false, false);
      }
    }
  }

  // ---- normalize + store to [B,T,C] ----
  #pragma unroll
  for (int r = 0; r < 8; ++r) {
    const float inv = 1.0f / lrun[r];
    const int q = qbase + r + 8 * hg;
    half_t* dst = AO + ((size_t)b * Tc + q) * Cc + h * Dc;
    #pragma unroll
    for (int ni = 0; ni < 8; ++ni)
      dst[ni * 16 + lm] = (half_t)(o[ni][r] * inv);
  }
}

// ---------------------------------------------------------------------------
extern "C" void kernel_launch(void* const* d_in, const int* in_sizes, int n_in,
                              void* d_out, int out_size, void* d_ws, size_t ws_size,
                              hipStream_t stream)
{
  (void)in_sizes; (void)n_in; (void)out_size; (void)ws_size;
  const float* x  = (const float*)d_in[0];
  const float* Wq = (const float*)d_in[1];
  const float* bq = (const float*)d_in[2];
  const float* Wk = (const float*)d_in[3];
  const float* bk = (const float*)d_in[4];
  const float* Wv = (const float*)d_in[5];
  const float* bv = (const float*)d_in[6];
  const float* Wo = (const float*)d_in[7];
  const float* bo = (const float*)d_in[8];
  float* out = (float*)d_out;

  const size_t headElems = (size_t)Bc * Hc * Tc * Dc;   // 8,388,608
  half_t* qbuf  = (half_t*)d_ws;
  half_t* kbuf  = qbuf + headElems;
  half_t* vbuf  = kbuf + headElems;
  half_t* aobuf = vbuf + headElems;   // [B,T,C] f16

  dim3 blk(256);
  dim3 gGemm(Cc / 128, Mc / 128);     // (16, 32)

  gemm_nt_bias<false, true><<<gGemm, blk, 0, stream>>>(x, Wq, bq, qbuf);
  gemm_nt_bias<false, true><<<gGemm, blk, 0, stream>>>(x, Wk, bk, kbuf);
  gemm_nt_bias<false, true><<<gGemm, blk, 0, stream>>>(x, Wv, bv, vbuf);

  const int ropeTotal = Bc * Hc * Tc * (Dc / 2);        // 4,194,304
  rope_qk<<<ropeTotal / 256, blk, 0, stream>>>(qbuf, kbuf);

  dim3 gAttn(Tc / 128, Bc * Hc);                        // (16, 32)
  flash_attn<<<gAttn, blk, 0, stream>>>(qbuf, kbuf, vbuf, aobuf);

  gemm_nt_bias<true, false><<<gGemm, blk, 0, stream>>>(aobuf, Wo, bo, out);
}
